// Encoder_7438883356782
// MI455X (gfx1250) — compile-verified, matrix operand order checked by reading, not run on hardware
//
#include <hip/hip_runtime.h>
#include <hip/hip_bf16.h>
#include <math.h>

// ---------------------------------------------------------------------------
// Pyraformer-style encoder for MI455X (gfx1250).
// GEMMs: LDS-tiled (128x128x32) v_wmma_f32_16x16x32_bf16, f32 accumulation,
//        global_prefetch of next K-tile.
// Attention: analytic pyramid mask -> sparse (<=10 keys/query) gather attention.
// ---------------------------------------------------------------------------

typedef __bf16 bf16;
typedef __attribute__((ext_vector_type(16))) __bf16 bf16x16;
typedef __attribute__((ext_vector_type(8)))  __bf16 bf16x8;
typedef __attribute__((ext_vector_type(8)))  float  f32x8;

#define BATCH   8
#define SEQL    1024
#define DM      512
#define DFF     2048
#define NH      8
#define DH      64
#define DBOT    128
#define LTOT    1360     // 1024 + 256 + 64 + 16
#define LALL    336      // 256 + 64 + 16
#define NROWS   (BATCH * LTOT)   // 10880

// ======================= WMMA GEMM: C[M,N] = A[M,K] * W[N,K]^T ==============
// Block 256 threads = 8 waves; tile 128(M) x 128(N) x 32(K) staged in LDS.
// Wave (mw = w&3, nw = w>>2) computes a 32(M) x 64(N) strip: 8 accumulators.
// LDS rows padded to 40 bf16 (80B) so 16-row fragment reads spread the banks.
#define BM 128
#define BN 128
#define BK 32
#define LPITCH 40

__device__ __forceinline__ bf16x16 make_frag(const bf16* row, int half)
{
    bf16x8 lo = *(const bf16x8*)(row + half * 8);
    bf16x8 hi = *(const bf16x8*)(row + 16 + half * 8);
    bf16x16 f;
#pragma unroll
    for (int e = 0; e < 8; ++e) { f[e] = lo[e]; f[8 + e] = hi[e]; }
    return f;
}

__global__ __launch_bounds__(256)
void gemm_wmma_bf16(const bf16* __restrict__ A, const bf16* __restrict__ W,
                    const float* __restrict__ bias,
                    float* __restrict__ Cf, bf16* __restrict__ Cb,
                    int M, int N, int K, int do_gelu)
{
    __shared__ bf16 As[BM * LPITCH];
    __shared__ bf16 Bs[BN * LPITCH];

    const int tid  = threadIdx.x;
    const int lane = tid & 31;
    const int wave = tid >> 5;
    const int half = lane >> 4;
    const int r    = lane & 15;
    const int mw   = wave & 3;       // 0..3 -> 32-row strip
    const int nw   = wave >> 2;      // 0..1 -> 64-col strip
    const int mBase = blockIdx.y * BM;
    const int nBase = blockIdx.x * BN;

    // staging assignment: thread handles chunks 2*tid, 2*tid+1 of 512
    const int c0   = tid * 2;
    const int row0 = c0 >> 2;            // 0..127
    const int cc0  = (c0 & 3) * 8;       // bf16 offset of first 16B chunk
    const int prow = tid & 127;          // prefetch row

    f32x8 acc[8] = {};

    for (int k0 = 0; k0 < K; k0 += BK) {
        // ---- global -> LDS staging (2x16B of A, 2x16B of B per thread) ----
        const bf16* ga = A + (size_t)(mBase + row0) * K + k0 + cc0;
        const bf16* gb = W + (size_t)(nBase + row0) * K + k0 + cc0;
        bf16* la = &As[row0 * LPITCH + cc0];
        bf16* lb = &Bs[row0 * LPITCH + cc0];
        *(bf16x8*)(la)     = *(const bf16x8*)(ga);
        *(bf16x8*)(la + 8) = *(const bf16x8*)(ga + 8);
        *(bf16x8*)(lb)     = *(const bf16x8*)(gb);
        *(bf16x8*)(lb + 8) = *(const bf16x8*)(gb + 8);

        // ---- prefetch next K-tile while this tile is consumed ----
        if (k0 + BK < K) {
            if (tid < 128)
                __builtin_prefetch(A + (size_t)(mBase + prow) * K + k0 + BK, 0, 1);
            else
                __builtin_prefetch(W + (size_t)(nBase + prow) * K + k0 + BK, 0, 1);
        }
        __syncthreads();

        // ---- compute: 8 WMMAs per wave on the LDS tile ----
        bf16x16 af0 = make_frag(&As[(mw * 32 + r) * LPITCH], half);
        bf16x16 af1 = make_frag(&As[(mw * 32 + 16 + r) * LPITCH], half);
#pragma unroll
        for (int j = 0; j < 4; ++j) {
            bf16x16 bfr = make_frag(&Bs[(nw * 64 + j * 16 + r) * LPITCH], half);
            acc[j * 2 + 0] = __builtin_amdgcn_wmma_f32_16x16x32_bf16(
                false, af0, false, bfr, (short)0, acc[j * 2 + 0], false, false);
            acc[j * 2 + 1] = __builtin_amdgcn_wmma_f32_16x16x32_bf16(
                false, af1, false, bfr, (short)0, acc[j * 2 + 1], false, false);
        }
        __syncthreads();
    }

    // ---- epilogue: bias (+GELU), f32 and/or bf16 stores ----
#pragma unroll
    for (int j = 0; j < 4; ++j) {
        const int col = nBase + nw * 64 + j * 16 + r;
        const float bv = bias ? bias[col] : 0.0f;
#pragma unroll
        for (int s = 0; s < 2; ++s) {
#pragma unroll
            for (int i = 0; i < 8; ++i) {
                const int row = mBase + mw * 32 + s * 16 + half * 8 + i;
                float v = acc[j * 2 + s][i] + bv;
                if (do_gelu) v = 0.5f * v * (1.0f + erff(v * 0.70710678118654752f));
                const size_t idx = (size_t)row * N + col;
                if (Cf) Cf[idx] = v;
                if (Cb) Cb[idx] = (bf16)v;
            }
        }
    }
}

// ======================= elementwise f32 -> bf16 ============================
__global__ void f32_to_bf16(const float* __restrict__ in, bf16* __restrict__ out, int n)
{
    int i = blockIdx.x * blockDim.x + threadIdx.x;
    if (i < n) out[i] = (bf16)in[i];
}

// ======================= token conv + pos + time embedding ==================
__global__ void embed_kernel(const float* __restrict__ xe, const float* __restrict__ xm,
                             const float* __restrict__ cw, const float* __restrict__ tw,
                             float* __restrict__ emb)
{
    int idx = blockIdx.x * blockDim.x + threadIdx.x;   // (b*1024+t)*512 + o
    if (idx >= BATCH * SEQL * DM) return;
    int o  = idx & (DM - 1);
    int bt = idx >> 9;
    int t  = bt & (SEQL - 1);
    int b  = bt >> 10;

    float acc = 0.0f;
#pragma unroll
    for (int kk = 0; kk < 3; ++kk) {
        int tt = (t + kk - 1 + SEQL) & (SEQL - 1);          // circular pad
        const float* xp = xe + (size_t)(b * SEQL + tt) * 7;
        const float* wp = cw + (size_t)o * 7 * 3 + kk;
#pragma unroll
        for (int c = 0; c < 7; ++c) acc += xp[c] * wp[c * 3];
    }
    int m2 = o & ~1;
    float freq = expf(-(float)m2 * (9.210340371976184f / (float)DM)); // ln(1e4)
    float ang  = (float)t * freq;
    acc += (o & 1) ? cosf(ang) : sinf(ang);
    const float* mp  = xm + (size_t)bt * 4;
    const float* twp = tw + (size_t)o * 4;
    acc += mp[0]*twp[0] + mp[1]*twp[1] + mp[2]*twp[2] + mp[3]*twp[3];
    emb[idx] = acc;
}

// ======================= (B,L,128) -> (B,128,L) transpose ===================
__global__ void transpose_blc(const float* __restrict__ in, float* __restrict__ out)
{
    int idx = blockIdx.x * blockDim.x + threadIdx.x;     // over 8*1024*128
    if (idx >= BATCH * SEQL * DBOT) return;
    int c = idx & (DBOT - 1);
    int l = (idx >> 7) & (SEQL - 1);
    int b = idx >> 17;
    out[((size_t)b * DBOT + c) * SEQL + l] = in[idx];
}

// ======================= CSCM strided conv (stride=4,k=4)+BN-scale+ELU ======
__global__ void cscm_conv(const float* __restrict__ tin, const float* __restrict__ w,
                          const float* __restrict__ bias, const float* __restrict__ g,
                          const float* __restrict__ beta,
                          float* __restrict__ tout, bf16* __restrict__ allin,
                          int Lin, int Lout, int off)
{
    int idx = blockIdx.x * blockDim.x + threadIdx.x;
    if (idx >= BATCH * DBOT * Lout) return;
    int l = idx % Lout;
    int o = (idx / Lout) & (DBOT - 1);
    int b = idx / (Lout * DBOT);

    const float* ip = tin + (size_t)b * DBOT * Lin + l * 4;
    const float* wp = w + (size_t)o * DBOT * 4;
    float acc = 0.0f;
    for (int c = 0; c < DBOT; ++c) {
        const float* ic = ip + (size_t)c * Lin;
        acc += ic[0]*wp[0] + ic[1]*wp[1] + ic[2]*wp[2] + ic[3]*wp[3];
        wp += 4;
    }
    acc += bias[o];
    acc = acc * 0.9999950000374997f * g[o] + beta[o];   // 1/sqrt(1+1e-5)
    acc = (acc > 0.0f) ? acc : (expf(acc) - 1.0f);      // ELU
    tout[((size_t)b * DBOT + o) * Lout + l] = acc;
    allin[((size_t)(b * LALL + off + l)) * DBOT + o] = (bf16)acc;
}

// ======================= build seq = concat(emb, up_out) ====================
__global__ void assemble_seq(const float* __restrict__ emb, const float* __restrict__ up,
                             float* __restrict__ outp)
{
    int idx = blockIdx.x * blockDim.x + threadIdx.x;     // over 8*1360*512
    if (idx >= NROWS * DM) return;
    int c = idx & (DM - 1);
    int p = (idx >> 9) % LTOT;
    int b = idx / (LTOT * DM);
    float v = (p < SEQL) ? emb[((size_t)b * SEQL + p) * DM + c]
                         : up[((size_t)b * LALL + (p - SEQL)) * DM + c];
    outp[idx] = v;
}

// ======================= LayerNorm (1 wave / 512-col row) ===================
__global__ __launch_bounds__(256)
void layernorm_rows(const float* __restrict__ X, const float* __restrict__ R,
                    const float* __restrict__ g, const float* __restrict__ bta,
                    float eps, float* __restrict__ Of, bf16* __restrict__ Ob, int rows)
{
    int wid  = (blockIdx.x * blockDim.x + threadIdx.x) >> 5;
    int lane = threadIdx.x & 31;
    if (wid >= rows) return;
    const float* x  = X + (size_t)wid * DM;
    const float* rr = R ? (R + (size_t)wid * DM) : nullptr;

    float v[16];
    float s = 0.0f;
#pragma unroll
    for (int t = 0; t < 16; ++t) {
        int c = lane + t * 32;
        float u = x[c] + (rr ? rr[c] : 0.0f);
        v[t] = u; s += u;
    }
#pragma unroll
    for (int off = 16; off; off >>= 1) s += __shfl_xor(s, off, 32);
    float mu = s * (1.0f / (float)DM);
    float q = 0.0f;
#pragma unroll
    for (int t = 0; t < 16; ++t) { float d = v[t] - mu; q += d * d; }
#pragma unroll
    for (int off = 16; off; off >>= 1) q += __shfl_xor(q, off, 32);
    float rstd = rsqrtf(q * (1.0f / (float)DM) + eps);
#pragma unroll
    for (int t = 0; t < 16; ++t) {
        int c = lane + t * 32;
        float y = (v[t] - mu) * rstd * g[c] + bta[c];
        size_t idx = (size_t)wid * DM + c;
        if (Of) Of[idx] = y;
        if (Ob) Ob[idx] = (bf16)y;
    }
}

// ======================= sparse pyramid attention ===========================
// Mask structure: per query <=10 allowed keys (5 intra-scale, 4 children,
// 1 parent). One wave per (b,h,i); lanes split the 64-dim head.
__global__ __launch_bounds__(256)
void attn_sparse(const bf16* __restrict__ Q, const bf16* __restrict__ Kt,
                 const bf16* __restrict__ V, bf16* __restrict__ O)
{
    int wid  = (blockIdx.x * blockDim.x + threadIdx.x) >> 5;  // 0 .. 8*8*1360-1
    int lane = threadIdx.x & 31;
    int i = wid % LTOT;
    int h = (wid / LTOT) & (NH - 1);
    int b = wid / (LTOT * NH);

    const int bases[4] = { 0, 1024, 1280, 1344 };
    const int sizes[4] = { 1024, 256, 64, 16 };
    int lvl = (i < 1024) ? 0 : (i < 1280) ? 1 : (i < 1344) ? 2 : 3;
    int loc = i - bases[lvl];

    int js[10]; int cnt = 0;
#pragma unroll
    for (int d = -2; d <= 2; ++d) {
        int l2 = loc + d;
        if (l2 >= 0 && l2 < sizes[lvl]) js[cnt++] = bases[lvl] + l2;
    }
    if (lvl > 0) {
        int cb = bases[lvl - 1] + loc * 4;
#pragma unroll
        for (int t = 0; t < 4; ++t) js[cnt++] = cb + t;
    }
    if (lvl < 3) js[cnt++] = bases[lvl + 1] + (loc >> 2);

    const size_t rowQ = ((size_t)(b * LTOT + i)) * DM + h * DH;
    float q0 = (float)Q[rowQ + lane];
    float q1 = (float)Q[rowQ + lane + 32];

    float sc[10];
    for (int t = 0; t < cnt; ++t) {
        const bf16* kp = Kt + ((size_t)(b * LTOT + js[t])) * DM + h * DH;
        float p = q0 * (float)kp[lane] + q1 * (float)kp[lane + 32];
        for (int off = 16; off; off >>= 1) p += __shfl_xor(p, off, 32);
        sc[t] = p * 0.125f;                       // 1/sqrt(64)
    }
    float mx = -1e30f;
    for (int t = 0; t < cnt; ++t) mx = fmaxf(mx, sc[t]);
    float sum = 0.0f;
    for (int t = 0; t < cnt; ++t) { sc[t] = expf(sc[t] - mx); sum += sc[t]; }
    float inv = 1.0f / sum;

    float o0 = 0.0f, o1 = 0.0f;
    for (int t = 0; t < cnt; ++t) {
        const bf16* vp = V + ((size_t)(b * LTOT + js[t])) * DM + h * DH;
        float p = sc[t] * inv;
        o0 += p * (float)vp[lane];
        o1 += p * (float)vp[lane + 32];
    }
    bf16* op = O + rowQ;
    op[lane]      = (bf16)o0;
    op[lane + 32] = (bf16)o1;
}

// ======================= final gather via indexes ===========================
__global__ void gather_out(const float* __restrict__ seq, const int* __restrict__ idxs,
                           float* __restrict__ out)
{
    int id = blockIdx.x * blockDim.x + threadIdx.x;    // 8*1024*4*512
    if (id >= BATCH * SEQL * 4 * DM) return;
    int d = id & (DM - 1);
    int j = (id >> 9) & 3;
    int i = (id >> 11) & (SEQL - 1);
    int b = id >> 21;
    out[id] = seq[((size_t)b * LTOT + idxs[i * 4 + j]) * DM + d];
}

// ===========================================================================
static inline int cdiv(int a, int b) { return (a + b - 1) / b; }

extern "C" void kernel_launch(void* const* d_in, const int* in_sizes, int n_in,
                              void* d_out, int out_size, void* d_ws, size_t ws_size,
                              hipStream_t stream)
{
    // ---- input indices (setup_inputs dict order, depth-first) ----
    const float* x_enc  = (const float*)d_in[0];
    const float* x_mark = (const float*)d_in[1];
    const float* tokw   = (const float*)d_in[2];
    const float* timew  = (const float*)d_in[3];
    const float* down_w = (const float*)d_in[4];
    const float* down_b = (const float*)d_in[5];
    const float* conv_w[3]   = { (const float*)d_in[6],  (const float*)d_in[10], (const float*)d_in[14] };
    const float* conv_b[3]   = { (const float*)d_in[7],  (const float*)d_in[11], (const float*)d_in[15] };
    const float* conv_g[3]   = { (const float*)d_in[8],  (const float*)d_in[12], (const float*)d_in[16] };
    const float* conv_bt[3]  = { (const float*)d_in[9],  (const float*)d_in[13], (const float*)d_in[17] };
    const float* up_w   = (const float*)d_in[18];
    const float* up_b   = (const float*)d_in[19];
    const float* cln_g  = (const float*)d_in[20];
    const float* cln_b  = (const float*)d_in[21];
    #define LW(l,k) ((const float*)d_in[22 + 14*(l) + (k)])
    const int*   indexes = (const int*)d_in[79];        // mask (d_in[78]) recomputed analytically
    float* out = (float*)d_out;

    // ---- workspace carve-up ----
    char* wp = (char*)d_ws;
    auto alloc = [&](size_t bytes) -> void* {
        void* r = (void*)wp; wp += (bytes + 255) & ~(size_t)255; return r;
    };
    float* emb      = (float*)alloc((size_t)BATCH*SEQL*DM*4);
    bf16*  emb_bf   = (bf16* )alloc((size_t)BATCH*SEQL*DM*2);
    float* down_o   = (float*)alloc((size_t)BATCH*SEQL*DBOT*4);
    float* tb0      = (float*)alloc((size_t)BATCH*DBOT*1024*4);
    float* tb1      = (float*)alloc((size_t)BATCH*DBOT*256*4);
    float* tb2      = (float*)alloc((size_t)BATCH*DBOT*64*4);
    float* tb3      = (float*)alloc((size_t)BATCH*DBOT*16*4);
    bf16*  allin_bf = (bf16* )alloc((size_t)BATCH*LALL*DBOT*2);
    float* up_o     = (float*)alloc((size_t)BATCH*LALL*DM*4);
    float* seq_f    = (float*)alloc((size_t)NROWS*DM*4);
    bf16*  seq_bf   = (bf16* )alloc((size_t)NROWS*DM*2);
    bf16*  q_bf     = (bf16* )alloc((size_t)NROWS*DM*2);
    bf16*  k_bf     = (bf16* )alloc((size_t)NROWS*DM*2);
    bf16*  v_bf     = (bf16* )alloc((size_t)NROWS*DM*2);
    bf16*  at_bf    = (bf16* )alloc((size_t)NROWS*DM*2);
    float* x_f      = (float*)alloc((size_t)NROWS*DM*4);   // also used as seq_raw
    bf16*  x_bf     = (bf16* )alloc((size_t)NROWS*DM*2);
    bf16*  h_bf     = (bf16* )alloc((size_t)NROWS*DFF*2);
    float* ffn_f    = (float*)alloc((size_t)NROWS*DM*4);
    bf16*  wdown    = (bf16* )alloc((size_t)DBOT*DM*2);
    bf16*  wup      = (bf16* )alloc((size_t)DM*DBOT*2);
    bf16*  wlayer[4][6];
    for (int l = 0; l < 4; ++l) {
        for (int k = 0; k < 4; ++k) wlayer[l][k] = (bf16*)alloc((size_t)DM*DM*2);     // q,k,v,o
        wlayer[l][4] = (bf16*)alloc((size_t)DFF*DM*2);                                 // w1
        wlayer[l][5] = (bf16*)alloc((size_t)DM*DFF*2);                                 // w2
    }
    (void)ws_size; (void)n_in; (void)in_sizes; (void)out_size;

    const int TB = 256;
    auto conv_f = [&](const float* src, bf16* dst, int n) {
        f32_to_bf16<<<cdiv(n, TB), TB, 0, stream>>>(src, dst, n);
    };

    // ---- weight conversion f32 -> bf16 ----
    conv_f(down_w, wdown, DBOT * DM);
    conv_f(up_w,   wup,   DM * DBOT);
    for (int l = 0; l < 4; ++l) {
        conv_f(LW(l,0), wlayer[l][0], DM*DM);
        conv_f(LW(l,2), wlayer[l][1], DM*DM);
        conv_f(LW(l,4), wlayer[l][2], DM*DM);
        conv_f(LW(l,6), wlayer[l][3], DM*DM);
        conv_f(LW(l,8), wlayer[l][4], DFF*DM);
        conv_f(LW(l,10),wlayer[l][5], DM*DFF);
    }

    // ---- embedding ----
    embed_kernel<<<cdiv(BATCH*SEQL*DM, TB), TB, 0, stream>>>(x_enc, x_mark, tokw, timew, emb);
    conv_f(emb, emb_bf, BATCH*SEQL*DM);

    // ---- down projection: (8192,512)x(128,512)^T ----
    gemm_wmma_bf16<<<dim3(DBOT/BN, (BATCH*SEQL)/BM), 256, 0, stream>>>(
        emb_bf, wdown, down_b, down_o, nullptr, BATCH*SEQL, DBOT, DM, 0);
    transpose_blc<<<cdiv(BATCH*SEQL*DBOT, TB), TB, 0, stream>>>(down_o, tb0);

    // ---- CSCM conv pyramid ----
    cscm_conv<<<cdiv(BATCH*DBOT*256, TB), TB, 0, stream>>>(
        tb0, conv_w[0], conv_b[0], conv_g[0], conv_bt[0], tb1, allin_bf, 1024, 256, 0);
    cscm_conv<<<cdiv(BATCH*DBOT*64, TB), TB, 0, stream>>>(
        tb1, conv_w[1], conv_b[1], conv_g[1], conv_bt[1], tb2, allin_bf, 256, 64, 256);
    cscm_conv<<<cdiv(BATCH*DBOT*16, TB), TB, 0, stream>>>(
        tb2, conv_w[2], conv_b[2], conv_g[2], conv_bt[2], tb3, allin_bf, 64, 16, 320);

    // ---- up projection: (2688,128)x(512,128)^T ----
    gemm_wmma_bf16<<<dim3(DM/BN, (BATCH*LALL)/BM), 256, 0, stream>>>(
        allin_bf, wup, up_b, up_o, nullptr, BATCH*LALL, DM, DBOT, 0);

    // ---- seq = LN(concat(emb, up), eps=1e-5) ----
    assemble_seq<<<cdiv(NROWS*DM, TB), TB, 0, stream>>>(emb, up_o, x_f);
    layernorm_rows<<<cdiv(NROWS*32, TB), TB, 0, stream>>>(
        x_f, nullptr, cln_g, cln_b, 1e-5f, seq_f, seq_bf, NROWS);

    // ---- encoder layers ----
    const dim3 gqkv(DM/BN, NROWS/BM);   // (4, 85)
    for (int l = 0; l < 4; ++l) {
        gemm_wmma_bf16<<<gqkv, 256, 0, stream>>>(seq_bf, wlayer[l][0], LW(l,1),
                                                 nullptr, q_bf, NROWS, DM, DM, 0);
        gemm_wmma_bf16<<<gqkv, 256, 0, stream>>>(seq_bf, wlayer[l][1], LW(l,3),
                                                 nullptr, k_bf, NROWS, DM, DM, 0);
        gemm_wmma_bf16<<<gqkv, 256, 0, stream>>>(seq_bf, wlayer[l][2], LW(l,5),
                                                 nullptr, v_bf, NROWS, DM, DM, 0);
        // sparse pyramid attention: one wave per (b,h,i)
        attn_sparse<<<(BATCH*NH*LTOT*32)/256, 256, 0, stream>>>(q_bf, k_bf, v_bf, at_bf);
        // output projection (f32 for residual + bf16 for FFN input)
        gemm_wmma_bf16<<<gqkv, 256, 0, stream>>>(at_bf, wlayer[l][3], LW(l,7),
                                                 x_f, x_bf, NROWS, DM, DM, 0);
        // FFN
        gemm_wmma_bf16<<<dim3(DFF/BN, NROWS/BM), 256, 0, stream>>>(
            x_bf, wlayer[l][4], LW(l,9), nullptr, h_bf, NROWS, DFF, DM, 1);
        gemm_wmma_bf16<<<gqkv, 256, 0, stream>>>(
            h_bf, wlayer[l][5], LW(l,11), ffn_f, nullptr, NROWS, DM, DFF, 0);
        // LN(ffn + residual, eps=1e-6) -> next seq
        layernorm_rows<<<cdiv(NROWS*32, TB), TB, 0, stream>>>(
            ffn_f, x_f, LW(l,12), LW(l,13), 1e-6f, seq_f, seq_bf, NROWS);
    }

    // ---- gather output (8,1024,4*512) ----
    gather_out<<<cdiv(BATCH*SEQL*4*DM, TB), TB, 0, stream>>>(seq_f, indexes, out);
}
